// FeedForward_39926015983895
// MI455X (gfx1250) — compile-verified, matrix operand order checked by reading, not run on hardware
//
#include <hip/hip_runtime.h>
#include <hip/hip_bf16.h>
#include <math.h>

typedef __attribute__((ext_vector_type(16))) _Float16 v16h;
typedef __attribute__((ext_vector_type(8)))  _Float16 v8h;
typedef __attribute__((ext_vector_type(8)))  float    v8f;

#define EMB   1024
#define FF    4096
#define MTOK  8192   // 4 * 2048 tokens

// ---------------- deterministic absmean reduction ----------------
__global__ void absmean_partial_kernel(const float* __restrict__ w,
                                       float* __restrict__ part, int chunk) {
    __shared__ float red[256];
    const int t = threadIdx.x;
    const size_t base = (size_t)blockIdx.x * (size_t)chunk;
    float s = 0.f;
    for (int i = t; i < chunk; i += 256) s += fabsf(w[base + i]);
    red[t] = s;
    __syncthreads();
    for (int off = 128; off > 0; off >>= 1) {
        if (t < off) red[t] += red[t + off];
        __syncthreads();
    }
    if (t == 0) part[blockIdx.x] = red[0];
}

__global__ void absmean_final_kernel(const float* __restrict__ part,
                                     float* __restrict__ inv_scale,
                                     int n, float invN) {
    __shared__ float red[256];
    const int t = threadIdx.x;
    float s = 0.f;
    for (int i = t; i < n; i += 256) s += part[i];
    red[t] = s;
    __syncthreads();
    for (int off = 128; off > 0; off >>= 1) {
        if (t < off) red[t] += red[t + off];
        __syncthreads();
    }
    if (t == 0) {
        float sc = fmaxf(red[0] * invN, 1e-8f);
        inv_scale[0] = 1.0f / sc;
    }
}

// w_t = clip(round(w/scale), -1, 1) stored exactly as f16 (-1/0/+1)
__global__ void quantize_ternary_kernel(const float* __restrict__ w,
                                        _Float16* __restrict__ o,
                                        const float* __restrict__ inv_scale, int n) {
    const float inv = inv_scale[0];
    const int stride = gridDim.x * blockDim.x;
    for (int i = blockIdx.x * blockDim.x + threadIdx.x; i < n; i += stride) {
        float q = __builtin_rintf(w[i] * inv);      // RNE == jnp.round
        q = fminf(1.f, fmaxf(-1.f, q));
        o[i] = (_Float16)q;
    }
}

__global__ void f32_to_f16_kernel(const float* __restrict__ x,
                                  _Float16* __restrict__ o, int n) {
    const int stride = gridDim.x * blockDim.x;
    for (int i = blockIdx.x * blockDim.x + threadIdx.x; i < n; i += stride)
        o[i] = (_Float16)x[i];
}

// ---------------- WMMA GEMM: C[M,N] = A[M,K] * B[N,K]^T (+bias, opt GELU) ----
#define BM  128
#define BN  128
#define BK  32
#define LDT 40    // padded LDS row stride in halfs (80B -> balanced 2-way banks)

template<bool GELU, typename OutT>
__global__ __launch_bounds__(256) void gemm_wmma_kernel(
    const _Float16* __restrict__ A, const _Float16* __restrict__ B,
    const float* __restrict__ bias, OutT* __restrict__ C,
    int M, int N, int K)
{
    __shared__ _Float16 As[BM * LDT];
    __shared__ _Float16 Bs[BN * LDT];

    const int t     = threadIdx.x;
    const int lane  = t & 31;
    const int wave  = t >> 5;
    const int waveM = wave & 3;          // 4 wave-rows of 32
    const int waveN = wave >> 2;         // 2 wave-cols of 64
    const int r     = lane & 15;
    const int hs    = (lane >> 4) << 3;  // 0 or 8 halfs (K half-select)

    const int blockN = blockIdx.x * BN;
    const int blockM = blockIdx.y * BM;

    // cooperative global->LDS staging: thread t loads 32B of one tile row
    const int rowT = t >> 1;             // 0..127
    const int ckT  = (t & 1) << 4;       // 0 or 16 halfs

    const _Float16* gA = A + (size_t)(blockM + rowT) * (size_t)K + ckT;
    const _Float16* gB = B + (size_t)(blockN + rowT) * (size_t)K + ckT;

    v8f acc[2][4] = {};

    uint4 ra0 = *(const uint4*)(gA);
    uint4 ra1 = *(const uint4*)(gA + 8);
    uint4 rb0 = *(const uint4*)(gB);
    uint4 rb1 = *(const uint4*)(gB + 8);

    const int nIter = K / BK;
    for (int it = 0; it < nIter; ++it) {
        _Float16* sa = &As[rowT * LDT + ckT];
        *(uint4*)(sa)     = ra0;
        *(uint4*)(sa + 8) = ra1;
        _Float16* sb = &Bs[rowT * LDT + ckT];
        *(uint4*)(sb)     = rb0;
        *(uint4*)(sb + 8) = rb1;
        __syncthreads();

        if (it + 1 < nIter) {            // overlap next tile's VMEM with WMMA
            const _Float16* nA = gA + (size_t)(it + 1) * BK;
            const _Float16* nB = gB + (size_t)(it + 1) * BK;
            ra0 = *(const uint4*)(nA); ra1 = *(const uint4*)(nA + 8);
            rb0 = *(const uint4*)(nB); rb1 = *(const uint4*)(nB + 8);
        }

        v16h a[2], b[4];
        #pragma unroll
        for (int i = 0; i < 2; ++i) {
            const _Float16* p = &As[(waveM * 32 + i * 16 + r) * LDT];
            ((v8h*)&a[i])[0] = *(const v8h*)(p + hs);
            ((v8h*)&a[i])[1] = *(const v8h*)(p + 16 + hs);
        }
        #pragma unroll
        for (int j = 0; j < 4; ++j) {
            const _Float16* p = &Bs[(waveN * 64 + j * 16 + r) * LDT];
            ((v8h*)&b[j])[0] = *(const v8h*)(p + hs);
            ((v8h*)&b[j])[1] = *(const v8h*)(p + 16 + hs);
        }

        #pragma unroll
        for (int i = 0; i < 2; ++i)
            #pragma unroll
            for (int j = 0; j < 4; ++j)
                acc[i][j] = __builtin_amdgcn_wmma_f32_16x16x32_f16(
                    false, a[i], false, b[j], (short)0, acc[i][j], false, false);

        __syncthreads();
    }

    // epilogue: C/D layout — VGPR e, lanes0-15: (M=e,N=lane); lanes16-31: (M=8+e)
    #pragma unroll
    for (int i = 0; i < 2; ++i) {
        const int mBase = blockM + waveM * 32 + i * 16 + hs;  // hs == 8*(lane>>4)
        #pragma unroll
        for (int j = 0; j < 4; ++j) {
            const int n  = blockN + waveN * 64 + j * 16 + r;
            const float bv = bias[n];
            #pragma unroll
            for (int e = 0; e < 8; ++e) {
                float v = acc[i][j][e] + bv;
                if (GELU) v = 0.5f * v * (1.0f + erff(v * 0.70710678118654752f));
                C[(size_t)(mBase + e) * (size_t)N + n] = (OutT)v;
            }
        }
    }
}

// ---------------- launcher ----------------
extern "C" void kernel_launch(void* const* d_in, const int* in_sizes, int n_in,
                              void* d_out, int out_size, void* d_ws, size_t ws_size,
                              hipStream_t stream) {
    const float* x  = (const float*)d_in[0];   // [4,2048,1024]
    const float* W1 = (const float*)d_in[1];   // [4096,1024]
    const float* b1 = (const float*)d_in[2];   // [4096]
    const float* W2 = (const float*)d_in[3];   // [1024,4096]
    const float* b2 = (const float*)d_in[4];   // [1024]
    float* out = (float*)d_out;                // [4,2048,1024] f32

    char* ws = (char*)d_ws;
    float*    part1 = (float*)(ws + 0);                 // 1024 f32
    float*    part2 = (float*)(ws + 4096);              // 1024 f32
    float*    inv1  = (float*)(ws + 8192);
    float*    inv2  = inv1 + 1;
    _Float16* xh    = (_Float16*)(ws + 16384);                          // 16 MB
    _Float16* W1h   = xh  + (size_t)MTOK * EMB;                         // 8 MB
    _Float16* W2h   = W1h + (size_t)FF * EMB;                           // 8 MB
    _Float16* h     = W2h + (size_t)EMB * FF;                           // 64 MB

    const int NW = FF * EMB;  // 4194304 weights each

    absmean_partial_kernel<<<1024, 256, 0, stream>>>(W1, part1, NW / 1024);
    absmean_partial_kernel<<<1024, 256, 0, stream>>>(W2, part2, NW / 1024);
    absmean_final_kernel<<<1, 256, 0, stream>>>(part1, inv1, 1024, 1.0f / (float)NW);
    absmean_final_kernel<<<1, 256, 0, stream>>>(part2, inv2, 1024, 1.0f / (float)NW);

    quantize_ternary_kernel<<<1024, 256, 0, stream>>>(W1, W1h, inv1, NW);
    quantize_ternary_kernel<<<1024, 256, 0, stream>>>(W2, W2h, inv2, NW);
    f32_to_f16_kernel<<<2048, 256, 0, stream>>>(x, xh, MTOK * EMB);

    // GEMM1 + bias + exact GELU -> h (f16):  [8192,1024] x [4096,1024]^T
    gemm_wmma_kernel<true, _Float16>
        <<<dim3(FF / BN, MTOK / BM), 256, 0, stream>>>(xh, W1h, b1, h, MTOK, FF, EMB);
    // GEMM2 + bias -> out (f32):             [8192,4096] x [1024,4096]^T
    gemm_wmma_kernel<false, float>
        <<<dim3(EMB / BN, MTOK / BM), 256, 0, stream>>>(h, W2h, b2, out, MTOK, EMB, FF);
}
